// GNNStack_3882650436708
// MI455X (gfx1250) — matrix operand order repaired and checked by reading, not compiled, not run on hardware
//
#include <hip/hip_runtime.h>
#include <math.h>

#define NN   20000
#define PP   3
#define LLP  10
#define LNS  20
#define DD   128
#define EE   640000
#define ND   (NN * DD)

#define SXP  (3 * DD + 4)   // padded float row for gate accumulators (388)
#define SHB  (DD + 8)       // padded ushort row for bf16 hidden state (136)
#define SAP  648            // padded ushort row for GEMM A-tile (>= 640)

typedef __bf16 bf16_t;
typedef bf16_t v16bf __attribute__((ext_vector_type(16)));
typedef float  v8f   __attribute__((ext_vector_type(8)));

union BF16x16 { v16bf v; unsigned short u[16]; };

__device__ __forceinline__ unsigned short f2bf(float f) {
    unsigned int u = __float_as_uint(f);
    unsigned int r = (u + 0x7FFFu + ((u >> 16) & 1u)) >> 16;
    return (unsigned short)r;
}

__device__ __forceinline__ v8f wmma_bf16(v16bf a, v16bf b, v8f c) {
    return __builtin_amdgcn_wmma_f32_16x16x32_bf16(false, a, false, b, (short)0, c, false, false);
}

// ---------------------------------------------------------------- conversions
__global__ __launch_bounds__(256) void f32_to_bf16_k(const float* __restrict__ s,
                                                     unsigned short* __restrict__ d,
                                                     long n) {
    long i = (long)blockIdx.x * 256 + threadIdx.x;
    if (i < n) d[i] = f2bf(s[i]);
}

// ---------------------------------------------------------------- fused GRU
// One block = 16 sequences. 8 waves; wave w owns gate-column tiles 3w..3w+2 (of 24).
// h kept fp32 (recurrence) + bf16 shadow (WMMA A-fragments) in LDS.
__global__ __launch_bounds__(256) void gru_kernel(
    const unsigned short* __restrict__ emb_bf,   // [V][128] bf16
    const int* __restrict__ idx,
    const int* __restrict__ lens,
    const unsigned short* __restrict__ Wih,      // [384][128] bf16
    const unsigned short* __restrict__ Whh,      // [384][128] bf16
    const float* __restrict__ bih,
    const float* __restrict__ bhh,
    float* __restrict__ out,                     // [B][128]
    int n_nodes, int T, int mode)                // mode 0: LP (s = p*N+ni), 1: NS
{
    __shared__ float          sh_h[16][DD];
    __shared__ unsigned short sh_hb[16][SHB];
    __shared__ float          sh_x[16][SXP];
    __shared__ float          sh_p[16][SXP];

    const int tid  = threadIdx.x;
    const int wave = tid >> 5;
    const int lane = tid & 31;
    const int ln   = lane & 15;
    const int hi   = lane >> 4;
    const int s0   = blockIdx.x * 16;

    for (int i = tid; i < 16 * DD; i += 256) {
        int m = i >> 7, d = i & (DD - 1);
        sh_h[m][d]  = 0.f;
        sh_hb[m][d] = 0;
    }
    __syncthreads();

    const int s_row = s0 + ln;
    int tokbase;
    if (mode == 0) {
        int p  = s_row / n_nodes;
        int ni = s_row - p * n_nodes;
        tokbase = (ni * PP + p) * T;
    } else {
        tokbase = s_row * T;
    }

    // Hoisted per-lane biases for this lane's 3 gate columns.
    float bx3[3], bh3[3];
#pragma unroll
    for (int j = 0; j < 3; ++j) {
        int col = (wave * 3 + j) * 16 + ln;
        bx3[j] = bih[col];
        bh3[j] = bhh[col];
    }
    // Hoisted per-thread sequence lengths for the 8 update elements.
    int mylen[8];
#pragma unroll
    for (int k = 0; k < 8; ++k) {
        int e = tid + k * 256;
        int s = s0 + (e >> 7);
        mylen[k] = (mode == 0) ? lens[(s % n_nodes) * PP + (s / n_nodes)] : lens[s];
    }

    for (int t = 0; t < T; ++t) {
        // A fragments: x_t from gathered bf16 embedding row, h from bf16 shadow.
        int tok = idx[tokbase + t];
        const unsigned short* er = emb_bf + (size_t)tok * DD;
        BF16x16 ax[4], ah[4];
#pragma unroll
        for (int kt = 0; kt < 4; ++kt) {
            int k0 = kt * 32 + hi * 8;
#pragma unroll
            for (int i = 0; i < 8; ++i) ax[kt].u[i]     = er[k0 + i];
#pragma unroll
            for (int i = 0; i < 8; ++i) ax[kt].u[8 + i] = er[k0 + 16 + i];
#pragma unroll
            for (int i = 0; i < 8; ++i) ah[kt].u[i]     = sh_hb[ln][k0 + i];
#pragma unroll
            for (int i = 0; i < 8; ++i) ah[kt].u[8 + i] = sh_hb[ln][k0 + 16 + i];
        }
#pragma unroll
        for (int j = 0; j < 3; ++j) {
            int col = (wave * 3 + j) * 16 + ln;
            v8f accx, acch;
#pragma unroll
            for (int r = 0; r < 8; ++r) { accx[r] = 0.f; acch[r] = 0.f; }
#pragma unroll
            for (int kt = 0; kt < 4; ++kt) {
                BF16x16 b;
                const unsigned short* wr = Wih + (size_t)col * DD + kt * 32 + hi * 8;
#pragma unroll
                for (int i = 0; i < 16; ++i) b.u[i] = wr[i < 8 ? i : 8 + i];
                accx = wmma_bf16(ax[kt].v, b.v, accx);
                wr = Whh + (size_t)col * DD + kt * 32 + hi * 8;
#pragma unroll
                for (int i = 0; i < 16; ++i) b.u[i] = wr[i < 8 ? i : 8 + i];
                acch = wmma_bf16(ah[kt].v, b.v, acch);
            }
#pragma unroll
            for (int r = 0; r < 8; ++r) {
                sh_x[r + hi * 8][col] = accx[r] + bx3[j];
                sh_p[r + hi * 8][col] = acch[r] + bh3[j];
            }
        }
        __syncthreads();
        // Elementwise gate update with pack_padded masking.
#pragma unroll
        for (int k = 0; k < 8; ++k) {
            int e = tid + k * 256;
            int m = e >> 7, d = e & (DD - 1);
            float xr = sh_x[m][d], xz = sh_x[m][DD + d], xn = sh_x[m][2 * DD + d];
            float hr = sh_p[m][d], hz = sh_p[m][DD + d], hn = sh_p[m][2 * DD + d];
            float r  = 1.f / (1.f + expf(-(xr + hr)));
            float z  = 1.f / (1.f + expf(-(xz + hz)));
            float nn = tanhf(xn + r * hn);
            float ho = sh_h[m][d];
            float hv = (1.f - z) * nn + z * ho;
            float hm = (t < mylen[k]) ? hv : ho;
            sh_h[m][d]  = hm;
            sh_hb[m][d] = f2bf(hm);
        }
        __syncthreads();
    }
    for (int k = 0; k < 8; ++k) {
        int e = tid + k * 256;
        int m = e >> 7, d = e & (DD - 1);
        out[(size_t)(s0 + m) * DD + d] = sh_h[m][d];
    }
}

// ---------------------------------------------------------------- generic GEMM (Nc = 128)
// C[M,128] = A[M,K](f32 -> bf16 staged in LDS) @ Wb[128,K]^T (+ bias).
__global__ __launch_bounds__(256) void gemm128(
    const float* __restrict__ A, const unsigned short* __restrict__ Wb,
    const float* __restrict__ bias, float* __restrict__ C, int M, int K)
{
    __shared__ unsigned short sA[16][SAP];
    int tid = threadIdx.x, wave = tid >> 5, lane = tid & 31, ln = lane & 15, hi = lane >> 4;
    int row0 = blockIdx.x * 16;

    // Cooperative staged conversion of the 16xK A tile (coalesced).
    for (int r = 0; r < 16; ++r) {
        const float* ar = A + (size_t)(row0 + r) * K;
        for (int c = tid; c < K; c += 256) sA[r][c] = f2bf(ar[c]);
    }
    __syncthreads();

    v8f acc;
#pragma unroll
    for (int r = 0; r < 8; ++r) acc[r] = 0.f;
    const unsigned short* wrow = Wb + (size_t)(wave * 16 + ln) * K;
    int nk = K >> 5;
    for (int kt = 0; kt < nk; ++kt) {
        BF16x16 a, b;
        int k0 = kt * 32 + hi * 8;
#pragma unroll
        for (int i = 0; i < 8; ++i) a.u[i]     = sA[ln][k0 + i];
#pragma unroll
        for (int i = 0; i < 8; ++i) a.u[8 + i] = sA[ln][k0 + 16 + i];
        const unsigned short* wp = wrow + kt * 32 + hi * 8;
#pragma unroll
        for (int i = 0; i < 16; ++i) b.u[i] = wp[i < 8 ? i : 8 + i];
        acc = wmma_bf16(a.v, b.v, acc);
    }
    int col = wave * 16 + ln;
    float bv = bias ? bias[col] : 0.f;
#pragma unroll
    for (int r = 0; r < 8; ++r) {
        int row = row0 + r + hi * 8;
        if (row < M) C[(size_t)row * DD + col] = acc[r] + bv;
    }
}

// ---------------------------------------------------------------- GCN helpers
__global__ __launch_bounds__(256) void deg_init_k(float* __restrict__ deg, int n) {
    int i = blockIdx.x * 256 + threadIdx.x;
    if (i < n) deg[i] = 1.0f;   // self-loop weight
}
__global__ __launch_bounds__(256) void deg_scatter_k(const int* __restrict__ ei,
                                                     const float* __restrict__ w,
                                                     float* __restrict__ deg, int E) {
    int e = blockIdx.x * 256 + threadIdx.x;
    if (e < E) atomicAdd(&deg[ei[E + e]], w[e]);
}
__global__ __launch_bounds__(256) void dinv_k(const float* __restrict__ deg,
                                              float* __restrict__ dinv, int n) {
    int i = blockIdx.x * 256 + threadIdx.x;
    if (i < n) dinv[i] = rsqrtf(deg[i]);
}
__global__ __launch_bounds__(256) void agg_init_k(const float* __restrict__ xw,
                                                  const float* __restrict__ dinv,
                                                  const float* __restrict__ b,
                                                  float* __restrict__ out, int n) {
    long i = (long)blockIdx.x * 256 + threadIdx.x;
    if (i < (long)n * DD) {
        int m = (int)(i >> 7), d = (int)(i & (DD - 1));
        out[i] = dinv[m] * dinv[m] * xw[i] + b[d];
    }
}
__global__ __launch_bounds__(256) void agg_scatter_k(const int* __restrict__ ei,
                                                     const float* __restrict__ w,
                                                     const float* __restrict__ dinv,
                                                     const float* __restrict__ xw,
                                                     float* __restrict__ out, int E) {
    long g = (long)blockIdx.x * 256 + threadIdx.x;
    if (g >= (long)E * 32) return;
    int e = (int)(g >> 5);
    int q = (int)(g & 31) * 4;
    int s = ei[e], d = ei[E + e];
    float c = dinv[s] * w[e] * dinv[d];
    const float* xs = xw + (size_t)s * DD + q;
    float* o = out + (size_t)d * DD + q;
    atomicAdd(o + 0, c * xs[0]);
    atomicAdd(o + 1, c * xs[1]);
    atomicAdd(o + 2, c * xs[2]);
    atomicAdd(o + 3, c * xs[3]);
}

// ---------------------------------------------------------------- activations
__global__ __launch_bounds__(256) void relu_ln_k(const float* __restrict__ in,
                                                 const float* __restrict__ g,
                                                 const float* __restrict__ b,
                                                 float* __restrict__ outp, int n) {
    int row  = blockIdx.x * 8 + (threadIdx.x >> 5);
    int lane = threadIdx.x & 31;
    if (row >= n) return;
    const float* r = in + (size_t)row * DD;
    float v[4], s = 0.f;
#pragma unroll
    for (int i = 0; i < 4; ++i) { v[i] = fmaxf(r[lane * 4 + i], 0.f); s += v[i]; }
#pragma unroll
    for (int off = 16; off > 0; off >>= 1) s += __shfl_xor(s, off, 32);
    float mean = s * (1.f / DD), q = 0.f;
#pragma unroll
    for (int i = 0; i < 4; ++i) { float d0 = v[i] - mean; q += d0 * d0; }
#pragma unroll
    for (int off = 16; off > 0; off >>= 1) q += __shfl_xor(q, off, 32);
    float inv = rsqrtf(q * (1.f / DD) + 1e-5f);
    float* o = outp + (size_t)row * DD;
#pragma unroll
    for (int i = 0; i < 4; ++i) {
        int d0 = lane * 4 + i;
        o[d0] = (v[i] - mean) * inv * g[d0] + b[d0];
    }
}
__global__ __launch_bounds__(256) void relu_k(const float* __restrict__ in,
                                              float* __restrict__ out, long n) {
    long i = (long)blockIdx.x * 256 + threadIdx.x;
    if (i < n) out[i] = fmaxf(in[i], 0.f);
}

// ---------------------------------------------------------------- head: mp2 + log_softmax
__global__ __launch_bounds__(256) void head_k(const float* __restrict__ y,
                                              const float* __restrict__ W2,
                                              const float* __restrict__ b2,
                                              float* __restrict__ outp, int n) {
    int i = blockIdx.x * 256 + threadIdx.x;
    if (i >= n) return;
    const float* r = y + (size_t)i * DD;
    float z0 = b2[0], z1 = b2[1];
    for (int d = 0; d < DD; ++d) {
        z0 += r[d] * W2[d];
        z1 += r[d] * W2[DD + d];
    }
    float m = fmaxf(z0, z1);
    float l = m + logf(expf(z0 - m) + expf(z1 - m));
    outp[(size_t)i * 2 + 0] = z0 - l;
    outp[(size_t)i * 2 + 1] = z1 - l;
}

// ---------------------------------------------------------------- launcher
extern "C" void kernel_launch(void* const* d_in, const int* in_sizes, int n_in,
                              void* d_out, int out_size, void* d_ws, size_t ws_size,
                              hipStream_t stream) {
    (void)in_sizes; (void)n_in; (void)out_size; (void)ws_size;
    const int*   idx_lp  = (const int*)d_in[0];
    const int*   lp_len  = (const int*)d_in[1];
    const int*   idx_ns  = (const int*)d_in[2];
    const int*   ns_len  = (const int*)d_in[3];
    const float* x_ref   = (const float*)d_in[4];
    const float* x_def   = (const float*)d_in[5];
    const float* x_pdt   = (const float*)d_in[6];
    const int*   edge_i  = (const int*)d_in[7];
    const float* edge_w  = (const float*)d_in[8];
    const float* lp_emb  = (const float*)d_in[9];
    const float* ns_emb  = (const float*)d_in[10];
    const float* lp_Wih  = (const float*)d_in[11];
    const float* lp_Whh  = (const float*)d_in[12];
    const float* lp_bih  = (const float*)d_in[13];
    const float* lp_bhh  = (const float*)d_in[14];
    const float* ns_Wih  = (const float*)d_in[15];
    const float* ns_Whh  = (const float*)d_in[16];
    const float* ns_bih  = (const float*)d_in[17];
    const float* ns_bhh  = (const float*)d_in[18];
    const float* lpfc_W  = (const float*)d_in[19];
    const float* lpfc_b  = (const float*)d_in[20];
    const float* allfc_W = (const float*)d_in[21];
    const float* allfc_b = (const float*)d_in[22];
    const float* conv_W[3] = {(const float*)d_in[23], (const float*)d_in[25], (const float*)d_in[27]};
    const float* conv_b[3] = {(const float*)d_in[24], (const float*)d_in[26], (const float*)d_in[28]};
    const float* ln_g[2]   = {(const float*)d_in[29], (const float*)d_in[31]};
    const float* ln_b[2]   = {(const float*)d_in[30], (const float*)d_in[32]};
    const float* mp1_W  = (const float*)d_in[33];
    const float* mp1_b  = (const float*)d_in[34];
    const float* mp2_W  = (const float*)d_in[35];
    const float* mp2_b  = (const float*)d_in[36];

    // ---- workspace layout
    size_t off = 0;
    auto aus = [&](size_t n) { void* p = (char*)d_ws + off; off += ((n * 2 + 255) / 256) * 256; return (unsigned short*)p; };
    auto af  = [&](size_t n) { void* p = (char*)d_ws + off; off += ((n * 4 + 255) / 256) * 256; return (float*)p; };

    unsigned short* wb_lp_ih = aus(384 * 128);
    unsigned short* wb_lp_hh = aus(384 * 128);
    unsigned short* wb_ns_ih = aus(384 * 128);
    unsigned short* wb_ns_hh = aus(384 * 128);
    unsigned short* wb_lpfc  = aus(128 * 384);
    unsigned short* wb_allfc = aus(128 * 640);
    unsigned short* wb_c[3]  = {aus(128 * 128), aus(128 * 128), aus(128 * 128)};
    unsigned short* wb_mp1   = aus(128 * 128);
    unsigned short* eb_lp    = aus((size_t)5000 * 128);
    unsigned short* eb_ns    = aus((size_t)10000 * 128);
    float* h_lp = af((size_t)PP * NN * DD);  // [P,N,D] == [N,3D] flat
    float* C5   = af((size_t)5 * ND);        // [pdt|ref|def|lp|ns] == [N,5D] flat
    float* X0   = af((size_t)ND);
    float* XW   = af((size_t)ND);
    float* XA   = af((size_t)ND);
    float* deg  = af(NN);
    float* dinv = af(NN);

    auto cvt = [&](const float* s, unsigned short* d, long n) {
        f32_to_bf16_k<<<(int)((n + 255) / 256), 256, 0, stream>>>(s, d, n);
    };
    cvt(lp_Wih, wb_lp_ih, 384 * 128);
    cvt(lp_Whh, wb_lp_hh, 384 * 128);
    cvt(ns_Wih, wb_ns_ih, 384 * 128);
    cvt(ns_Whh, wb_ns_hh, 384 * 128);
    cvt(lpfc_W, wb_lpfc, 128 * 384);
    cvt(allfc_W, wb_allfc, 128 * 640);
    for (int i = 0; i < 3; ++i) cvt(conv_W[i], wb_c[i], 128 * 128);
    cvt(mp1_W, wb_mp1, 128 * 128);
    cvt(lp_emb, eb_lp, (long)5000 * 128);
    cvt(ns_emb, eb_ns, (long)10000 * 128);

    // faithful concat buffer sections 0..2
    hipMemcpyAsync(C5 + 0 * (size_t)ND, x_pdt, (size_t)ND * 4, hipMemcpyDeviceToDevice, stream);
    hipMemcpyAsync(C5 + 1 * (size_t)ND, x_ref, (size_t)ND * 4, hipMemcpyDeviceToDevice, stream);
    hipMemcpyAsync(C5 + 2 * (size_t)ND, x_def, (size_t)ND * 4, hipMemcpyDeviceToDevice, stream);

    // GRUs
    gru_kernel<<<(PP * NN) / 16, 256, 0, stream>>>(eb_lp, idx_lp, lp_len, wb_lp_ih, wb_lp_hh,
                                                   lp_bih, lp_bhh, h_lp, NN, LLP, 0);
    gru_kernel<<<NN / 16, 256, 0, stream>>>(eb_ns, idx_ns, ns_len, wb_ns_ih, wb_ns_hh,
                                            ns_bih, ns_bhh, C5 + 4 * (size_t)ND, NN, LNS, 1);

    // lp_fc ([N,384] -> [N,128]) and all_fc ([N,640] -> [N,128])
    gemm128<<<NN / 16, 256, 0, stream>>>(h_lp, wb_lpfc, lpfc_b, C5 + 3 * (size_t)ND, NN, 3 * DD);
    gemm128<<<NN / 16, 256, 0, stream>>>(C5, wb_allfc, allfc_b, X0, NN, 5 * DD);

    // symmetric normalization (shared by all convs)
    deg_init_k<<<(NN + 255) / 256, 256, 0, stream>>>(deg, NN);
    deg_scatter_k<<<(EE + 255) / 256, 256, 0, stream>>>(edge_i, edge_w, deg, EE);
    dinv_k<<<(NN + 255) / 256, 256, 0, stream>>>(deg, dinv, NN);

    float* emb_out  = (float*)d_out;
    float* head_out = (float*)d_out + (size_t)ND;
    for (int i = 0; i < 3; ++i) {
        gemm128<<<NN / 16, 256, 0, stream>>>(X0, wb_c[i], nullptr, XW, NN, DD);
        float* dst = (i == 2) ? emb_out : XA;
        agg_init_k<<<(int)(((long)ND + 255) / 256), 256, 0, stream>>>(XW, dinv, conv_b[i], dst, NN);
        long work = (long)EE * 32;
        agg_scatter_k<<<(int)((work + 255) / 256), 256, 0, stream>>>(edge_i, edge_w, dinv, XW, dst, EE);
        if (i < 2)
            relu_ln_k<<<(NN + 7) / 8, 256, 0, stream>>>(dst, ln_g[i], ln_b[i], X0, NN);
        else
            relu_k<<<(int)(((long)ND + 255) / 256), 256, 0, stream>>>(dst, X0, (long)ND);
    }

    // head
    gemm128<<<NN / 16, 256, 0, stream>>>(X0, wb_mp1, mp1_b, XW, NN, DD);
    head_k<<<(NN + 255) / 256, 256, 0, stream>>>(XW, mp2_W, mp2_b, head_out, NN);
}
